// BertMultiHeadSelfAttention_57604101374440
// MI455X (gfx1250) — matrix operand order repaired and checked by reading, not compiled, not run on hardware
//
#include <hip/hip_runtime.h>
#include <hip/hip_bf16.h>

// ---------------------------------------------------------------------------
// BERT multi-head self-attention for gfx1250 (MI455X), wave32 + WMMA bf16.
//   B=2, S=2048, H=768, NH=12, HD=64
// Pipeline: f32->bf16 convert, 3x WMMA GEMM (QKV) with TDM tile staging,
//           fused flash attention, WMMA GEMM output projection (fp32 out).
// ---------------------------------------------------------------------------

typedef __attribute__((ext_vector_type(16))) __bf16         v16bf;
typedef __attribute__((ext_vector_type(8)))  float          v8f;
typedef __attribute__((ext_vector_type(8)))  unsigned short v8u16;
typedef __attribute__((ext_vector_type(16))) unsigned short v16u16;
typedef __attribute__((ext_vector_type(4)))  unsigned int   v4u;
typedef __attribute__((ext_vector_type(8)))  int            v8i;
typedef __attribute__((ext_vector_type(4)))  int            v4i;

#if defined(__AMDGCN__) && __has_builtin(__builtin_amdgcn_tensor_load_to_lds) && \
    __has_builtin(__builtin_amdgcn_s_wait_tensorcnt)
#define USE_TDM 1
#else
#define USE_TDM 0
#endif

__device__ __forceinline__ unsigned short f2bf(float f) {
  unsigned int u = __float_as_uint(f);
  // round-to-nearest-even truncation to bf16
  return (unsigned short)((u + 0x7FFFu + ((u >> 16) & 1u)) >> 16);
}

__device__ __forceinline__ v16bf make_frag(v8u16 lo, v8u16 hi) {
  v16u16 w = __builtin_shufflevector(lo, hi, 0, 1, 2, 3, 4, 5, 6, 7,
                                     8, 9, 10, 11, 12, 13, 14, 15);
  return __builtin_bit_cast(v16bf, w);
}

__device__ __forceinline__ v16bf bfrag(const unsigned short* p, int ld) {
  v16u16 w;
#pragma unroll
  for (int s = 0; s < 8; ++s) {
    w[s] = p[s * ld];
    w[s + 8] = p[(16 + s) * ld];
  }
  return __builtin_bit_cast(v16bf, w);
}

#if USE_TDM
// Issue one TDM 2D tile load (bf16 elements) into LDS.
//   gsrc        : global address of the TILE start (= tensor base + offset)
//   ldsoff      : LDS byte offset of destination (generic-ptr low 32 bits)
//   tensor_d0/1 : tensor extent in elements (dim0 = fastest-varying)
//   tile_d0/1   : tile extent in elements
//   stride0     : elements between consecutive dim1 lines
//   pad_int/amt : D# LDS pad codes (interval 2^(c+1) DWORDs / amount c+1 DW)
__device__ __forceinline__ void tdm_load_tile_bf16(
    const unsigned short* gsrc, unsigned ldsoff, unsigned tensor_d0,
    unsigned tensor_d1, unsigned tile_d0, unsigned tile_d1, unsigned stride0,
    unsigned pad_int, unsigned pad_amt) {
  unsigned long long ga = (unsigned long long)(size_t)gsrc;
  // D# group0: count=1 | lds_addr | global_addr(57b) | type=2 ("image")
  v4u g0;
  g0.x = 1u;
  g0.y = ldsoff;
  g0.z = (unsigned)ga;
  g0.w = (unsigned)((ga >> 32) & 0x1FFFFFFu) | (2u << 30);
  // D# group1
  unsigned w0 = (1u << 16)                       // data_size = 2 bytes
                | (1u << 20)                     // pad_enable
                | (pad_int << 22) | (pad_amt << 25);
  v8i g1;
  g1[0] = (int)w0;                                           // wg_mask=0 etc.
  g1[1] = (int)((tensor_d0 & 0xFFFFu) << 16);                // abar=0|dim0.lo
  g1[2] = (int)((tensor_d0 >> 16) | ((tensor_d1 & 0xFFFFu) << 16));
  g1[3] = (int)((tensor_d1 >> 16) | (tile_d0 << 16));
  g1[4] = (int)(tile_d1);                                    // tile_dim2 = 0
  g1[5] = (int)stride0;                                      // dim0 stride lo
  g1[6] = 0;                                                 // hi | dim1str.lo
  g1[7] = 0;
  v4i z4 = (v4i){};
  v8i z8 = (v8i){};
  // clang-23 / therock form: (g0, g1, g2, g3, g4, cpol)
  __builtin_amdgcn_tensor_load_to_lds(g0, g1, z4, z4, z8, 0);
}
#endif

// ---------------------------------------------------------------------------
// fp32 -> bf16 conversion
// ---------------------------------------------------------------------------
__global__ void f32_to_bf16_kernel(const float* __restrict__ in,
                                   unsigned short* __restrict__ out, int n) {
  int i = blockIdx.x * blockDim.x + threadIdx.x;
  if (i < n) out[i] = f2bf(in[i]);
}

// ---------------------------------------------------------------------------
// C[M,N] = A[M,K](bf16) @ B[K,N](bf16) + bias   (out bf16 or fp32)
// Workgroup tile 128x128, BK=32, 8 waves (4x2), wave tile 32x64.
// Tiles staged in LDS by the Tensor Data Mover when available.
// ---------------------------------------------------------------------------
#define GA_STRIDE 40   // 128x32 A tile rows: 16 DW data + 4 DW pad
#define GB_STRIDE 136  // 32x128 B tile rows: 64 DW data + 4 DW pad

__global__ __launch_bounds__(256) void gemm_bf16_kernel(
    const unsigned short* __restrict__ A, const unsigned short* __restrict__ Bm,
    const float* __restrict__ bias, unsigned short* __restrict__ Cb,
    float* __restrict__ Cf, int M, int N, int K) {
  __shared__ unsigned short As[128 * GA_STRIDE];
  __shared__ unsigned short Bs[32 * GB_STRIDE];

  const int tid  = threadIdx.x;
  const int lane = tid & 31;
  const int w    = tid >> 5;
  const int ntiles = N >> 7;
  const int bx = blockIdx.x % ntiles;
  const int by = blockIdx.x / ntiles;
  const int m0 = by << 7;
  const int n0 = bx << 7;
  const int wm = w & 3;   // 4 wave rows of 32
  const int wn = w >> 2;  // 2 wave cols of 64

  v8f acc[2][4];
#pragma unroll
  for (int i = 0; i < 2; ++i)
#pragma unroll
    for (int j = 0; j < 4; ++j) acc[i][j] = (v8f){};

  const int arow = lane & 15;
  const int kb = (lane < 16) ? 0 : 8;

#if USE_TDM
  const unsigned as_off = (unsigned)(size_t)(const void*)&As[0];
  const unsigned bs_off = (unsigned)(size_t)(const void*)&Bs[0];
#else
  const int ar = tid >> 2, ac = (tid & 3) << 3;   // A loader: 64 rows/round
  const int br = tid >> 4, bc = (tid & 15) << 3;  // B loader: 16 rows/round
#endif

  for (int kt = 0; kt < K; kt += 32) {
#if USE_TDM
    if (w == 0) {
      // A tile: 128 rows x 32 (dim0 = K direction), rows padded to 40 elems
      tdm_load_tile_bf16(A + (size_t)m0 * K + kt, as_off, (unsigned)K,
                         (unsigned)M, 32u, 128u, (unsigned)K, 3u, 3u);
      // B tile: 32 rows x 128 (dim0 = N direction), rows padded to 136 elems
      tdm_load_tile_bf16(Bm + (size_t)kt * N + n0, bs_off, (unsigned)N,
                         (unsigned)K, 128u, 32u, (unsigned)N, 5u, 3u);
      __builtin_amdgcn_s_wait_tensorcnt(0);
    }
#else
#pragma unroll
    for (int r2 = 0; r2 < 2; ++r2) {
      int row = ar + (r2 << 6);
      *(v8u16*)&As[row * GA_STRIDE + ac] =
          *(const v8u16*)&A[(size_t)(m0 + row) * K + kt + ac];
    }
#pragma unroll
    for (int r2 = 0; r2 < 2; ++r2) {
      int row = br + (r2 << 4);
      *(v8u16*)&Bs[row * GB_STRIDE + bc] =
          *(const v8u16*)&Bm[(size_t)(kt + row) * N + n0 + bc];
    }
#endif
    __syncthreads();

    v16bf afr[2], bfr[4];
#pragma unroll
    for (int i = 0; i < 2; ++i) {
      const unsigned short* p = &As[(wm * 32 + i * 16 + arow) * GA_STRIDE + kb];
      afr[i] = make_frag(*(const v8u16*)p, *(const v8u16*)(p + 16));
    }
#pragma unroll
    for (int j = 0; j < 4; ++j)
      bfr[j] = bfrag(&Bs[kb * GB_STRIDE + wn * 64 + j * 16 + arow], GB_STRIDE);

#pragma unroll
    for (int i = 0; i < 2; ++i)
#pragma unroll
      for (int j = 0; j < 4; ++j)
        acc[i][j] = __builtin_amdgcn_wmma_f32_16x16x32_bf16(
            false, afr[i], false, bfr[j], (short)0, acc[i][j], false, false);
    __syncthreads();
  }

  // Epilogue: branch once on output type; incremental addressing.
  const int rowext = (lane < 16) ? 0 : 8;
  const int crow = m0 + wm * 32 + rowext;
  const int ccol = n0 + wn * 64 + (lane & 15);
  float bv[4];
#pragma unroll
  for (int j = 0; j < 4; ++j) bv[j] = bias[ccol + j * 16];

  if (Cf != nullptr) {
#pragma unroll
    for (int i = 0; i < 2; ++i) {
#pragma unroll
      for (int j = 0; j < 4; ++j) {
        float* p = Cf + (size_t)(crow + i * 16) * N + ccol + j * 16;
#pragma unroll
        for (int r = 0; r < 8; ++r) p[(size_t)r * N] = acc[i][j][r] + bv[j];
      }
    }
  } else {
#pragma unroll
    for (int i = 0; i < 2; ++i) {
#pragma unroll
      for (int j = 0; j < 4; ++j) {
        unsigned short* p = Cb + (size_t)(crow + i * 16) * N + ccol + j * 16;
#pragma unroll
        for (int r = 0; r < 8; ++r) p[(size_t)r * N] = f2bf(acc[i][j][r] + bv[j]);
      }
    }
  }
}

// ---------------------------------------------------------------------------
// Fused flash attention: grid = B*NH*(S/64) blocks of 128 threads (4 waves).
// Each wave handles 16 query rows; kv streamed in tiles of 32.
// Q/K/V/ctx stored bf16 as [B, S, H] with head h at column h*64.
// ---------------------------------------------------------------------------
#define KT_STRIDE 40  // transposed K tile [64 d][32 n], padded
#define PS_STRIDE 40  // per-wave P staging [16][32], padded
#define LOG2E 1.442695040888963f

__global__ __launch_bounds__(128) void attn_kernel(
    const unsigned short* __restrict__ Qm, const unsigned short* __restrict__ Km,
    const unsigned short* __restrict__ Vm, const int* __restrict__ amask,
    unsigned short* __restrict__ ctx) {
  __shared__ unsigned short Kt[64 * KT_STRIDE];
  __shared__ unsigned short Pst[4][16 * PS_STRIDE];

  const int tid  = threadIdx.x;
  const int lane = tid & 31;
  const int w    = tid >> 5;
  const int blk  = blockIdx.x;
  const int qt = blk & 31;  // 32 q-tiles of 64
  const int bh = blk >> 5;  // 0..23
  const int b = bh / 12;
  const int h = bh % 12;
  const int qbase = qt * 64 + w * 16;
  const int arow = lane & 15;
  const int kb = (lane < 16) ? 0 : 8;
  const int rowext = (lane < 16) ? 0 : 8;

  const size_t headoff = (size_t)h * 64;
  const unsigned short* Qh = Qm + (size_t)b * 2048 * 768 + headoff;
  const unsigned short* Kh = Km + (size_t)b * 2048 * 768 + headoff;
  const unsigned short* Vh = Vm + (size_t)b * 2048 * 768 + headoff;

  // Q fragments for both d-steps (d0 = 0, 32), loaded once from global.
  v16bf qf[2];
#pragma unroll
  for (int t = 0; t < 2; ++t) {
    const unsigned short* p = Qh + (size_t)(qbase + arow) * 768 + t * 32 + kb;
    qf[t] = make_frag(*(const v8u16*)p, *(const v8u16*)(p + 16));
  }

  v8f oacc[4];
#pragma unroll
  for (int t = 0; t < 4; ++t) oacc[t] = (v8f){};
  float mi[8], li[8];
#pragma unroll
  for (int r = 0; r < 8; ++r) { mi[r] = -1e30f; li[r] = 0.0f; }

  const int kn = tid >> 3;        // kv row loader: 16 rows per round
  const int kd = (tid & 7) << 3;  // d-chunk of 8

  for (int n0 = 0; n0 < 2048; n0 += 32) {
    // Stage K tile transposed into LDS: Kt[d][n] = K[n0+n][d]
#pragma unroll
    for (int r2 = 0; r2 < 2; ++r2) {
      int n = kn + r2 * 16;
      v8u16 kv = *(const v8u16*)(Kh + (size_t)(n0 + n) * 768 + kd);
#pragma unroll
      for (int e = 0; e < 8; ++e) Kt[(kd + e) * KT_STRIDE + n] = kv[e];
    }
    // Prefetch next K tile into cache (global_prefetch_b8)
    {
      int nn = (n0 + 32 < 2048) ? (n0 + 32) : n0;
      __builtin_prefetch(Kh + (size_t)(nn + kn) * 768 + kd, 0, 1);
    }
    __syncthreads();

    // S = Q @ K^T for two 16-wide n-subtiles, K-dim 64 in two WMMA steps
    v8f s0 = (v8f){}, s1 = (v8f){};
#pragma unroll
    for (int t = 0; t < 2; ++t) {
      const unsigned short* pb = &Kt[(t * 32 + kb) * KT_STRIDE + arow];
      s0 = __builtin_amdgcn_wmma_f32_16x16x32_bf16(false, qf[t], false,
                                                   bfrag(pb, KT_STRIDE), (short)0, s0,
                                                   false, false);
      s1 = __builtin_amdgcn_wmma_f32_16x16x32_bf16(false, qf[t], false,
                                                   bfrag(pb + 16, KT_STRIDE), (short)0, s1,
                                                   false, false);
    }

    // attention-mask adder for this lane's two kv columns
    int c0 = n0 + (lane & 15);
    float ad0 = (1.0f - (float)amask[b * 2048 + c0]) * -10000.0f;
    float ad1 = (1.0f - (float)amask[b * 2048 + c0 + 16]) * -10000.0f;

    // Online softmax; VGPR r <-> query row (r or r+8 per half-wave).
    unsigned short* Pw = &Pst[w][0];
#pragma unroll
    for (int r = 0; r < 8; ++r) {
      float x0 = s0[r] * 0.125f + ad0;
      float x1 = s1[r] * 0.125f + ad1;
      float mx = fmaxf(x0, x1);
#pragma unroll
      for (int off = 1; off < 16; off <<= 1)
        mx = fmaxf(mx, __shfl_xor(mx, off, 16));
      float mn = fmaxf(mi[r], mx);
      float corr = exp2f((mi[r] - mn) * LOG2E);
      float e0 = exp2f((x0 - mn) * LOG2E);
      float e1 = exp2f((x1 - mn) * LOG2E);
      float rs = e0 + e1;
#pragma unroll
      for (int off = 1; off < 16; off <<= 1) rs += __shfl_xor(rs, off, 16);
      li[r] = li[r] * corr + rs;
      mi[r] = mn;
#pragma unroll
      for (int t = 0; t < 4; ++t) oacc[t][r] *= corr;
      int prow = r + rowext;
      Pw[prow * PS_STRIDE + (lane & 15)] = f2bf(e0);
      Pw[prow * PS_STRIDE + 16 + (lane & 15)] = f2bf(e1);
    }

    // P: D-layout -> A-layout via wave-private LDS (in-order per wave).
    const unsigned short* pp = &Pw[arow * PS_STRIDE + kb];
    v16bf pf = make_frag(*(const v8u16*)pp, *(const v8u16*)(pp + 16));

    // O += P @ V ; V B-fragments straight from global (lane = d column, elements = kv rows).
#pragma unroll
    for (int t = 0; t < 4; ++t) {
      const unsigned short* vp = Vh + (size_t)(n0 + kb) * 768 + t * 16 + arow;
      oacc[t] = __builtin_amdgcn_wmma_f32_16x16x32_bf16(
          false, pf, false, bfrag(vp, 768), (short)0, oacc[t], false, false);
    }
    __syncthreads();
  }

  // Normalize and write ctx (bf16, [B,S,H] layout)
#pragma unroll
  for (int t = 0; t < 4; ++t) {
#pragma unroll
    for (int r = 0; r < 8; ++r) {
      int row = qbase + r + rowext;
      float v = oacc[t][r] / li[r];
      ctx[((size_t)b * 2048 + row) * 768 + headoff + t * 16 + (lane & 15)] =
          f2bf(v);
    }
  }
}

// ---------------------------------------------------------------------------
// Host launcher
// ---------------------------------------------------------------------------
extern "C" void kernel_launch(void* const* d_in, const int* in_sizes, int n_in,
                              void* d_out, int out_size, void* d_ws,
                              size_t ws_size, hipStream_t stream) {
  const float* hs   = (const float*)d_in[0];
  const int* amask  = (const int*)d_in[1];
  const float* Wq   = (const float*)d_in[2];
  const float* bq   = (const float*)d_in[3];
  const float* Wk   = (const float*)d_in[4];
  const float* bk   = (const float*)d_in[5];
  const float* Wv   = (const float*)d_in[6];
  const float* bv   = (const float*)d_in[7];
  const float* Wo   = (const float*)d_in[8];
  const float* bo   = (const float*)d_in[9];
  float* out = (float*)d_out;

  const size_t nHS = (size_t)2 * 2048 * 768;  // 3,145,728
  const size_t nW  = (size_t)768 * 768;       // 589,824

  unsigned short* ws    = (unsigned short*)d_ws;
  unsigned short* hs_bf = ws;
  unsigned short* wq_bf = hs_bf + nHS;
  unsigned short* wk_bf = wq_bf + nW;
  unsigned short* wv_bf = wk_bf + nW;
  unsigned short* wo_bf = wv_bf + nW;
  unsigned short* q_bf  = wo_bf + nW;
  unsigned short* k_bf  = q_bf + nHS;
  unsigned short* v_bf  = k_bf + nHS;
  unsigned short* c_bf  = v_bf + nHS;

  auto cvt = [&](const float* src, unsigned short* dst, size_t n) {
    f32_to_bf16_kernel<<<(unsigned)((n + 511) / 512), 512, 0, stream>>>(
        src, dst, (int)n);
  };
  cvt(hs, hs_bf, nHS);
  cvt(Wq, wq_bf, nW);
  cvt(Wk, wk_bf, nW);
  cvt(Wv, wv_bf, nW);
  cvt(Wo, wo_bf, nW);

  // QKV projections: M=4096, N=768, K=768 -> (768/128)*(4096/128) = 192 blocks
  dim3 gg(192), gb(256);
  gemm_bf16_kernel<<<gg, gb, 0, stream>>>(hs_bf, wq_bf, bq, q_bf, nullptr,
                                          4096, 768, 768);
  gemm_bf16_kernel<<<gg, gb, 0, stream>>>(hs_bf, wk_bf, bk, k_bf, nullptr,
                                          4096, 768, 768);
  gemm_bf16_kernel<<<gg, gb, 0, stream>>>(hs_bf, wv_bf, bv, v_bf, nullptr,
                                          4096, 768, 768);

  // Fused attention: B*NH*(S/64) = 2*12*32 = 768 blocks of 128 threads
  attn_kernel<<<768, 128, 0, stream>>>(q_bf, k_bf, v_bf, amask, c_bf);

  // Output projection -> fp32 d_out
  gemm_bf16_kernel<<<gg, gb, 0, stream>>>(c_bf, wo_bf, bo, nullptr, out, 4096,
                                          768, 768);
}
